// EvoformerStack_72842645340696
// MI455X (gfx1250) — matrix operand order repaired and checked by reading, not compiled, not run on hardware
//
#include <hip/hip_runtime.h>

// ---------------------------------------------------------------------------
// Evoformer (2 blocks) for MI455X / gfx1250.
// Every contraction -> v_wmma_f32_16x16x32_bf16 (bf16 operands, fp32 accum).
// All GEMMs are NT (B pre-transposed / pre-permuted), so both operands are
// K-contiguous and fragment loads are pure global_load_b128. The K-loop is
// software-pipelined (double-buffered fragments) and versioned on the
// second-subtile predicate so the hot loop is branch-free.
// ---------------------------------------------------------------------------

typedef unsigned short bf16raw;
typedef __attribute__((ext_vector_type(16))) __bf16 v16bf;
typedef __attribute__((ext_vector_type(8)))  float  v8f;

__device__ __forceinline__ bf16raw f2bfr(float f) {
  unsigned u = __builtin_bit_cast(unsigned, f);
  unsigned r = u + 0x7FFFu + ((u >> 16) & 1u);   // round-to-nearest-even
  return (bf16raw)(r >> 16);
}

// two contiguous 16-byte chunks -> one 16-element bf16 fragment
__device__ __forceinline__ v16bf load_frag16(const bf16raw* p0, const bf16raw* p1) {
  union { uint4 q[2]; v16bf v; } u;
  u.q[0] = *(const uint4*)p0;
  u.q[1] = *(const uint4*)p1;
  return u.v;
}

// ---------------------------------------------------------------------------
// Batched NT WMMA GEMM:  C[b] = epi( scale * A[b] (MxK, lda) x B[b]^T (NxK, ldb) )
//   batch b = blockIdx.z ; b1 = b / nb2 ; b2 = b % nb2 ; off = b1*s?1 + b2*s?2
//   epilogue: +bias[n], act(0 none,1 relu,2 sigmoid), *mul, +res, fp32/bf16 out
// Block = 256 threads = 8 waves; wave w owns rows [w*16,+16) of a 128-row
// macro-tile and a 16x32 output strip (2 WMMAs / K-step, shared A fragment).
// ---------------------------------------------------------------------------
__global__ void k_gemm(const bf16raw* __restrict__ A,
                       const bf16raw* __restrict__ B,
                       void* __restrict__ Cv,
                       const float* __restrict__ bias,
                       const float* __restrict__ res,
                       const float* __restrict__ mul,
                       int M, int N, int K,
                       long lda, long ldb, long ldc,
                       long sA1, long sA2, long sB1, long sB2, long sC1, long sC2,
                       int nb2, float scale, int act, int outBf) {
  int wave = threadIdx.x >> 5;
  int lane = threadIdx.x & 31;
  int row0 = blockIdx.y * 128 + wave * 16;
  int col0 = blockIdx.x * 32;
  if (row0 >= M || col0 >= N) return;          // wave-uniform: EXEC stays full
  bool t1 = (col0 + 16) < N;                   // second 16-col subtile active?

  long b  = blockIdx.z;
  long b1 = b / nb2;
  long b2 = b - b1 * (long)nb2;
  const bf16raw* Ab = A + b1 * sA1 + b2 * sA2;
  const bf16raw* Bb = B + b1 * sB1 + b2 * sB2;
  long coff = b1 * sC1 + b2 * sC2;

  int g = lane >> 4;        // half-wave group
  int l = lane & 15;
  int am = row0 + l;
  int bn0 = col0 + l;       if (bn0 >= N) bn0 = N - 1;   // clamp tiny-N heads
  int bn1 = col0 + 16 + l;  if (bn1 >= N) bn1 = N - 1;

  // ISA 16-bit A 16x32 layout: lane chunks at k = g*8 and k = 16+g*8
  // ISA 16-bit B 32x16 layout: lane chunks at k = g*8 and k = 16+g*8 (same as A)
  const bf16raw* arow  = Ab + (long)am * lda + g * 8;
  const bf16raw* brow0 = Bb + (long)bn0 * ldb + g * 8;
  const bf16raw* brow1 = Bb + (long)bn1 * ldb + g * 8;

  v8f acc0 = {}, acc1 = {};
  if (t1) {
    // ---- two-subtile pipelined loop (branch-free body) ----
    v16bf af = load_frag16(arow, arow + 16);
    v16bf b0 = load_frag16(brow0, brow0 + 16);
    v16bf b1f = load_frag16(brow1, brow1 + 16);
    for (int k0 = 32; k0 < K; k0 += 32) {
      v16bf afn = load_frag16(arow + k0, arow + k0 + 16);
      v16bf b0n = load_frag16(brow0 + k0, brow0 + k0 + 16);
      v16bf b1n = load_frag16(brow1 + k0, brow1 + k0 + 16);
      acc0 = __builtin_amdgcn_wmma_f32_16x16x32_bf16(false, af, false, b0,
                                                     (short)0, acc0, false, false);
      acc1 = __builtin_amdgcn_wmma_f32_16x16x32_bf16(false, af, false, b1f,
                                                     (short)0, acc1, false, false);
      af = afn; b0 = b0n; b1f = b1n;
    }
    acc0 = __builtin_amdgcn_wmma_f32_16x16x32_bf16(false, af, false, b0,
                                                   (short)0, acc0, false, false);
    acc1 = __builtin_amdgcn_wmma_f32_16x16x32_bf16(false, af, false, b1f,
                                                   (short)0, acc1, false, false);
  } else {
    // ---- single-subtile pipelined loop ----
    v16bf af = load_frag16(arow, arow + 16);
    v16bf b0 = load_frag16(brow0, brow0 + 16);
    for (int k0 = 32; k0 < K; k0 += 32) {
      v16bf afn = load_frag16(arow + k0, arow + k0 + 16);
      v16bf b0n = load_frag16(brow0 + k0, brow0 + k0 + 16);
      acc0 = __builtin_amdgcn_wmma_f32_16x16x32_bf16(false, af, false, b0,
                                                     (short)0, acc0, false, false);
      af = afn; b0 = b0n;
    }
    acc0 = __builtin_amdgcn_wmma_f32_16x16x32_bf16(false, af, false, b0,
                                                   (short)0, acc0, false, false);
  }

#pragma unroll
  for (int t = 0; t < 2; ++t) {
    if (t == 1 && !t1) break;
    v8f acc = t ? acc1 : acc0;
#pragma unroll
    for (int v = 0; v < 8; ++v) {
      int m = row0 + v + 8 * g;              // C/D layout: VGPR v -> M=v (+8 hi)
      int n = col0 + t * 16 + l;
      if (n >= N) continue;
      float val = acc[v] * scale;
      if (bias) val += bias[n];
      if (act == 1) val = val > 0.f ? val : 0.f;
      else if (act == 2) val = 1.f / (1.f + __expf(-val));
      long cidx = coff + (long)m * ldc + n;
      if (mul) val *= mul[cidx];
      if (res) val += res[cidx];
      if (outBf) ((bf16raw*)Cv)[cidx] = f2bfr(val);
      else       ((float*)Cv)[cidx]   = val;
    }
  }
}

// ---------------------------------------------------------------------------
// LayerNorm over last dim d, fp32 in -> bf16 out (LN outputs only feed GEMMs).
// ---------------------------------------------------------------------------
__global__ void k_layernorm(const float* __restrict__ x, bf16raw* __restrict__ y,
                            const float* __restrict__ gam, const float* __restrict__ bet,
                            int d) {
  long row = blockIdx.x;
  const float* xr = x + row * (long)d;
  __shared__ float sm[128];
  int tid = threadIdx.x;
  float s = 0.f;
  for (int i = tid; i < d; i += 128) s += xr[i];
  sm[tid] = s; __syncthreads();
  for (int o = 64; o; o >>= 1) { if (tid < o) sm[tid] += sm[tid + o]; __syncthreads(); }
  float mean = sm[0] / d; __syncthreads();
  float v = 0.f;
  for (int i = tid; i < d; i += 128) { float t = xr[i] - mean; v += t * t; }
  sm[tid] = v; __syncthreads();
  for (int o = 64; o; o >>= 1) { if (tid < o) sm[tid] += sm[tid + o]; __syncthreads(); }
  float rstd = rsqrtf(sm[0] / d + 1e-5f);
  for (int i = tid; i < d; i += 128)
    y[row * (long)d + i] = f2bfr((xr[i] - mean) * rstd * gam[i] + bet[i]);
}

// ---------------------------------------------------------------------------
// Softmax over last dim Kd, additive bias at bias[(b%bmod)*sb + r*sr + k*sk].
// fp32 logits in -> bf16 probabilities out.
// ---------------------------------------------------------------------------
__global__ void k_softmax(const float* __restrict__ x, bf16raw* __restrict__ y,
                          const float* __restrict__ bias,
                          int rowsPerB, int Kd, int bmod, long sb, long sr, long sk) {
  long rid = blockIdx.x;
  long b = rid / rowsPerB;
  long r = rid - b * rowsPerB;
  const float* xr = x + rid * (long)Kd;
  const float* bp = bias ? bias + (b % bmod) * sb + r * sr : nullptr;
  __shared__ float sm[128];
  int tid = threadIdx.x;
  float mx = -3.4e38f;
  for (int i = tid; i < Kd; i += 128) {
    float v = xr[i] + (bp ? bp[(long)i * sk] : 0.f);
    mx = fmaxf(mx, v);
  }
  sm[tid] = mx; __syncthreads();
  for (int o = 64; o; o >>= 1) { if (tid < o) sm[tid] = fmaxf(sm[tid], sm[tid + o]); __syncthreads(); }
  float M = sm[0]; __syncthreads();
  float s = 0.f;
  for (int i = tid; i < Kd; i += 128) {
    float v = xr[i] + (bp ? bp[(long)i * sk] : 0.f);
    s += __expf(v - M);
  }
  sm[tid] = s; __syncthreads();
  for (int o = 64; o; o >>= 1) { if (tid < o) sm[tid] += sm[tid + o]; __syncthreads(); }
  float inv = 1.f / sm[0];
  for (int i = tid; i < Kd; i += 128) {
    float v = xr[i] + (bp ? bp[(long)i * sk] : 0.f);
    y[rid * (long)Kd + i] = f2bfr(__expf(v - M) * inv);
  }
}

// out[(i1,i2,h,c)] = bf16( gate[(i1,i2,h,c)] * ob[b, rr, c] )
//   mode 0: b=i1*H+h, rr=i2, rows=D2 ; mode 1: b=i2*H+h, rr=i1, rows=D1
__global__ void k_gate_mul(const float* __restrict__ ob, const float* __restrict__ gate,
                           bf16raw* __restrict__ out, int D1, int D2, int H, int C,
                           int mode) {
  long idx = blockIdx.x * 256L + threadIdx.x;
  long tot = (long)D1 * D2 * H * C;
  if (idx >= tot) return;
  int c = idx % C; long t = idx / C;
  int h = t % H; t /= H;
  int i2 = t % D2; long i1 = t / D2;
  long b = mode ? ((long)i2 * H + h) : (i1 * H + h);
  long rr = mode ? i1 : (long)i2;
  long rows = mode ? D1 : D2;
  float v = gate[idx] * ob[(b * rows + rr) * (long)C + c];
  out[idx] = f2bfr(v);
}

// V (d1,d2,H,C) bf16 -> VT[(b*C + c)*Klen + kk] bf16 so A.V becomes NT GEMM.
//   mode 0: b=x*H+h, kk=y, Klen=D2 ; mode 1: b=y*H+h, kk=x, Klen=D1
__global__ void k_vperm(const bf16raw* __restrict__ v, bf16raw* __restrict__ vt,
                        int D1, int D2, int H, int C, int mode) {
  long idx = blockIdx.x * 256L + threadIdx.x;
  long tot = (long)D1 * D2 * H * C;
  if (idx >= tot) return;
  int c = idx % C; long t = idx / C;
  int h = t % H; t /= H;
  int y = t % D2; long x = t / D2;
  long b = mode ? ((long)y * H + h) : (x * H + h);
  long kk = mode ? x : (long)y;
  long Klen = mode ? D1 : D2;
  vt[(b * (long)C + c) * Klen + kk] = v[idx];
}

// plain fp32 -> bf16 convert
__global__ void k_cvt(const float* __restrict__ s, bf16raw* __restrict__ d, long n) {
  long i = blockIdx.x * 256L + threadIdx.x;
  if (i < n) d[i] = f2bfr(s[i]);
}

// transpose-convert: src (rows, cols) fp32 -> dst (cols, rows) bf16
__global__ void k_cvt_t(const float* __restrict__ s, bf16raw* __restrict__ d,
                        int rows, int cols) {
  long i = blockIdx.x * 256L + threadIdx.x;
  long n = (long)rows * cols;
  if (i >= n) return;
  int c = i % cols; long r = i / cols;
  d[(long)c * rows + r] = f2bfr(s[i]);
}

// tri-mul pack: val = gmat*pmat at (x,y,c); dst[c][x][y] (swap=0) or dst[c][y][x]
__global__ void k_pack_gate(const float* __restrict__ gm, const float* __restrict__ pm,
                            bf16raw* __restrict__ dst, int R_, int Cc, int swap) {
  long idx = blockIdx.x * 256L + threadIdx.x;
  long tot = (long)R_ * R_ * Cc;
  if (idx >= tot) return;
  int c = idx % Cc; long t = idx / Cc;
  int y = t % R_; long x = t / R_;
  float v = gm[idx] * pm[idx];
  long o = (long)c * R_ * R_ + (swap ? (long)y * R_ + x : x * (long)R_ + y);
  dst[o] = f2bfr(v);
}

// (c,i,j) fp32 -> (i,j,c) fp32
__global__ void k_perm_cij_ijc(const float* __restrict__ s, float* __restrict__ d,
                               int R_, int Cc) {
  long idx = blockIdx.x * 256L + threadIdx.x;
  long tot = (long)R_ * R_ * Cc;
  if (idx >= tot) return;
  int j = idx % R_; long t = idx / R_;
  int i = t % R_; long c = t / R_;
  d[((long)i * R_ + j) * Cc + c] = s[idx];
}

// OPM reorder: src[(i*Cc+c), (j*Cc+dd)] fp32 -> dst[(i*R+j), c*Cc+dd] bf16
__global__ void k_opm_perm(const float* __restrict__ s, bf16raw* __restrict__ d,
                           int R_, int Cc) {
  long idx = blockIdx.x * 256L + threadIdx.x;
  long W = (long)R_ * Cc;
  long tot = W * W;
  if (idx >= tot) return;
  long col = idx % W, row = idx / W;
  int i = row / Cc, c = row % Cc;
  int j = col / Cc, dd = col % Cc;
  d[((long)i * R_ + j) * (long)(Cc * Cc) + c * Cc + dd] = f2bfr(s[idx]);
}

// ---------------------------------------------------------------------------
// Host orchestration
// ---------------------------------------------------------------------------
struct PLin { const float* w; const float* b; };
struct PLN  { const float* g; const float* b; };
struct PAttn { PLN ln, ln_z; PLin q, k, v, g, o, bias; };
struct PTriMul { PLN ln, ln_out; PLin ap, ag, bp, bg, g, o; };
struct PTrans { PLN ln; PLin w1, w2; };
struct POPM { PLN ln; PLin a, b, o; };
struct PBlock { PAttn row, col, tas, tae; PTrans mt, pt; POPM opm; PTriMul tmo, tmi; };

static void gemm(hipStream_t st, const bf16raw* A, const bf16raw* B, void* C,
                 const float* bias, const float* res, const float* mul,
                 int M, int N, int K, long lda, long ldb, long ldc,
                 int nBtot, int nb2,
                 long sA1, long sA2, long sB1, long sB2, long sC1, long sC2,
                 float scale, int act, int outBf) {
  dim3 gr((N + 31) / 32, (M + 127) / 128, nBtot);
  k_gemm<<<gr, 256, 0, st>>>(A, B, C, bias, res, mul, M, N, K, lda, ldb, ldc,
                             sA1, sA2, sB1, sB2, sC1, sC2, nb2, scale, act, outBf);
}

extern "C" void kernel_launch(void* const* d_in, const int* in_sizes, int n_in,
                              void* d_out, int out_size, void* d_ws, size_t ws_size,
                              hipStream_t stream) {
  (void)in_sizes; (void)out_size;
  const int S = 128, R = 192, CM = 256, CZ = 128, CS = 384;
  const int HM = 8, HT = 4, CH = 32;
  const long SR = (long)S * R;      // 24576
  const long RR = (long)R * R;      // 36864
  const float rsq32 = 0.17677669529663688f;   // 1/sqrt(32)

  if (n_in < 186) return;
  const float* in_m = (const float*)d_in[0];
  const float* in_z = (const float*)d_in[1];

  // ---- parse params (jax tree-flatten: dict keys sorted) ----
  int ci = 2;
  auto F    = [&]() { return (const float*)d_in[ci++]; };
  auto linB = [&](PLin& l) { l.b = F(); l.w = F(); };          // "b","w"
  auto linN = [&](PLin& l) { l.b = nullptr; l.w = F(); };      // "w"
  auto lnp  = [&](PLN& l)  { l.b = F(); l.g = F(); };          // "b","g"
  auto attnNoPair = [&](PAttn& a) {   // g,k,ln,o,q,v
    linB(a.g); linN(a.k); lnp(a.ln); linB(a.o); linN(a.q); linN(a.v);
    a.bias.w = nullptr; a.bias.b = nullptr;
  };
  auto attnPair = [&](PAttn& a) {     // bias,g,k,ln,ln_z,o,q,v
    linN(a.bias); linB(a.g); linN(a.k); lnp(a.ln); lnp(a.ln_z);
    linB(a.o); linN(a.q); linN(a.v);
  };
  auto attnTri = [&](PAttn& a) {      // bias,g,k,ln,o,q,v
    linN(a.bias); linB(a.g); linN(a.k); lnp(a.ln);
    linB(a.o); linN(a.q); linN(a.v);
  };
  auto transp = [&](PTrans& t) { lnp(t.ln); linB(t.w1); linB(t.w2); };
  auto opmp   = [&](POPM& o) { linN(o.a); linN(o.b); lnp(o.ln); linB(o.o); };
  auto trimul = [&](PTriMul& t) {     // ag,ap,bg,bp,g,ln,ln_out,o
    linB(t.ag); linB(t.ap); linB(t.bg); linB(t.bp);
    linB(t.g); lnp(t.ln); lnp(t.ln_out); linB(t.o);
  };
  PBlock blk[2];
  for (int i = 0; i < 2; ++i) {       // block keys: col,mt,opm,pt,row,tae,tas,tmi,tmo
    attnNoPair(blk[i].col); transp(blk[i].mt); opmp(blk[i].opm); transp(blk[i].pt);
    attnPair(blk[i].row); attnTri(blk[i].tae); attnTri(blk[i].tas);
    trimul(blk[i].tmi); trimul(blk[i].tmo);
  }
  PLin fin; linB(fin);

  // ---- outputs live in d_out ----
  float* m_cur = (float*)d_out;           // SR*CM
  float* z_cur = m_cur + SR * CM;         // RR*CZ
  float* s_out = z_cur + RR * CZ;         // R*CS

  // ---- workspace carve ----
  char* wp = (char*)d_ws;
  auto carve = [&](long elems, int esz) -> void* {
    void* p = wp; wp += ((long)elems * esz + 255) & ~255L; return p;
  };
  bf16raw* MLN = (bf16raw*)carve(SR * CM, 2);       // LN(m) bf16 / bf16 scratch
  bf16raw* ZLN = (bf16raw*)carve(RR * CZ, 2);       // LN(z) bf16
  bf16raw* WB  = (bf16raw*)carve(256 * 1024, 2);    // transposed weight bf16
  bf16raw* QB  = (bf16raw*)carve(SR * CM, 2);
  bf16raw* KB  = (bf16raw*)carve(SR * CM, 2);
  bf16raw* VB  = (bf16raw*)carve(SR * CM, 2);
  bf16raw* VT  = (bf16raw*)carve(SR * CM, 2);       // per-(batch,head) V^T
  float* GT    = (float*)carve(SR * CM, 4);         // sigmoid gate fp32
  float* BUF1  = (float*)carve(SR * CM, 4);
  float* BUF2  = (float*)carve(SR * CM, 4);
  float* LOG   = (float*)carve((long)S * HM * RR, 4);   // logits / OPM outer
  bf16raw* ATT = (bf16raw*)carve((long)S * HM * RR, 2); // softmax / OPM bf16
  bf16raw* GOB = (bf16raw*)carve(SR * CM, 2);       // gated attention out bf16
  bf16raw* H1  = (bf16raw*)carve(SR * 1024, 2);     // transition hidden bf16
  float* PB    = (float*)carve(RR * HM, 4);         // pair bias
  bf16raw* AT  = (bf16raw*)carve(RR * CZ, 2);       // tri-mul packed A
  bf16raw* BT  = (bf16raw*)carve(RR * CZ, 2);       // tri-mul packed B
  float* X1    = (float*)carve(RR * CZ, 4);         // tri-mul (c,i,j)
  float* X2    = (float*)carve(RR * CZ, 4);         // tri-mul (i,j,c)
  float* GMAT  = (float*)carve(RR * CZ, 4);         // tri-mul output gate
  if ((size_t)(wp - (char*)d_ws) > ws_size) return;

  hipMemcpyAsync(m_cur, in_m, SR * CM * sizeof(float), hipMemcpyDeviceToDevice, stream);
  hipMemcpyAsync(z_cur, in_z, RR * CZ * sizeof(float), hipMemcpyDeviceToDevice, stream);

  auto g1 = [](long n) { return dim3((unsigned)((n + 255) / 256)); };
  auto LN = [&](const float* x, bf16raw* y, const PLN& p, long rows, int d) {
    k_layernorm<<<dim3((unsigned)rows), 128, 0, stream>>>(x, y, p.g, p.b, d);
  };
  auto CVT_T = [&](const float* w, int din, int dout) {
    k_cvt_t<<<g1((long)din * dout), 256, 0, stream>>>(w, WB, din, dout);
  };
  // X(rows,din) @ W(din,dout): W transposed to bf16, NT GEMM
  auto PROJ = [&](const bf16raw* Xln, const PLin& L, int din, int dout, long rows,
                  void* out, int act, int outBf, const float* res, const float* mulp) {
    CVT_T(L.w, din, dout);
    gemm(stream, Xln, WB, out, L.b, res, mulp, (int)rows, dout, din,
         din, din, dout, 1, 1, 0, 0, 0, 0, 0, 0, 1.f, act, outBf);
  };

  for (int bi = 0; bi < 2; ++bi) {
    const PBlock& B = blk[bi];

    // ================= MSA row attention (pair bias) =================
    LN(m_cur, MLN, B.row.ln, SR, CM);
    LN(z_cur, ZLN, B.row.ln_z, RR, CZ);
    PROJ(MLN, B.row.q, CM, CM, SR, QB, 0, 1, 0, 0);
    PROJ(MLN, B.row.k, CM, CM, SR, KB, 0, 1, 0, 0);
    PROJ(MLN, B.row.v, CM, CM, SR, VB, 0, 1, 0, 0);
    PROJ(MLN, B.row.g, CM, CM, SR, GT, 2, 0, 0, 0);
    PROJ(ZLN, B.row.bias, CZ, HM, RR, PB, 0, 0, 0, 0);
    k_vperm<<<g1(SR * CM), 256, 0, stream>>>(VB, VT, S, R, HM, CH, 0);
    // logits[s,h,r,k] = q.k^T/sqrt(c) ; batch (s,h)
    gemm(stream, QB, KB, LOG, 0, 0, 0, R, R, CH, CM, CM, R,
         S * HM, HM, (long)R * CM, CH, (long)R * CM, CH, (long)HM * RR, RR,
         rsq32, 0, 0);
    // + bias[r*R*H + k*H + h], softmax over k
    k_softmax<<<dim3((unsigned)(S * HM * R)), 128, 0, stream>>>(
        LOG, ATT, PB, R, R, HM, 1L, (long)R * HM, (long)HM);
    // o[b,r,c] = A.V^T ; batch (s,h)
    gemm(stream, ATT, VT, BUF1, 0, 0, 0, R, CH, R, R, R, CH,
         S * HM, HM, (long)HM * RR, RR, (long)HM * CH * R, (long)CH * R,
         (long)HM * R * CH, (long)R * CH, 1.f, 0, 0);
    k_gate_mul<<<g1(SR * CM), 256, 0, stream>>>(BUF1, GT, GOB, S, R, HM, CH, 0);
    CVT_T(B.row.o.w, CM, CM);
    gemm(stream, GOB, WB, m_cur, B.row.o.b, m_cur, 0, (int)SR, CM, CM,
         CM, CM, CM, 1, 1, 0, 0, 0, 0, 0, 0, 1.f, 0, 0);

    // ================= MSA column attention =================
    LN(m_cur, MLN, B.col.ln, SR, CM);
    PROJ(MLN, B.col.q, CM, CM, SR, QB, 0, 1, 0, 0);
    PROJ(MLN, B.col.k, CM, CM, SR, KB, 0, 1, 0, 0);
    PROJ(MLN, B.col.v, CM, CM, SR, VB, 0, 1, 0, 0);
    PROJ(MLN, B.col.g, CM, CM, SR, GT, 2, 0, 0, 0);
    k_vperm<<<g1(SR * CM), 256, 0, stream>>>(VB, VT, S, R, HM, CH, 1);
    // logits[r,h,s,t] ; batch (r,h); A row stride over s = R*CM
    gemm(stream, QB, KB, LOG, 0, 0, 0, S, S, CH, (long)R * CM, (long)R * CM, S,
         R * HM, HM, CM, CH, CM, CH, (long)HM * S * S, (long)S * S,
         rsq32, 0, 0);
    k_softmax<<<dim3((unsigned)(R * HM * S)), 128, 0, stream>>>(
        LOG, ATT, nullptr, S, S, 1, 0L, 0L, 0L);
    gemm(stream, ATT, VT, BUF1, 0, 0, 0, S, CH, S, S, S, CH,
         R * HM, HM, (long)HM * S * S, (long)S * S, (long)HM * CH * S, (long)CH * S,
         (long)HM * S * CH, (long)S * CH, 1.f, 0, 0);
    k_gate_mul<<<g1(SR * CM), 256, 0, stream>>>(BUF1, GT, GOB, S, R, HM, CH, 1);
    CVT_T(B.col.o.w, CM, CM);
    gemm(stream, GOB, WB, m_cur, B.col.o.b, m_cur, 0, (int)SR, CM, CM,
         CM, CM, CM, 1, 1, 0, 0, 0, 0, 0, 0, 1.f, 0, 0);

    // ================= MSA transition =================
    LN(m_cur, MLN, B.mt.ln, SR, CM);
    CVT_T(B.mt.w1.w, CM, 4 * CM);
    gemm(stream, MLN, WB, H1, B.mt.w1.b, 0, 0, (int)SR, 4 * CM, CM,
         CM, CM, 4 * CM, 1, 1, 0, 0, 0, 0, 0, 0, 1.f, 1, 1);   // relu, bf16 out
    CVT_T(B.mt.w2.w, 4 * CM, CM);
    gemm(stream, H1, WB, m_cur, B.mt.w2.b, m_cur, 0, (int)SR, CM, 4 * CM,
         4 * CM, 4 * CM, CM, 1, 1, 0, 0, 0, 0, 0, 0, 1.f, 0, 0);

    // ================= Outer product mean =================
    LN(m_cur, MLN, B.opm.ln, SR, CM);
    PROJ(MLN, B.opm.a, CM, 32, SR, BUF1, 0, 0, 0, 0);   // a (S,R,32) fp32
    PROJ(MLN, B.opm.b, CM, 32, SR, BUF2, 0, 0, 0, 0);
    k_cvt_t<<<g1(SR * 32), 256, 0, stream>>>(BUF1, AT, S, R * 32);  // (R*32, S)
    k_cvt_t<<<g1(SR * 32), 256, 0, stream>>>(BUF2, BT, S, R * 32);
    gemm(stream, AT, BT, LOG, 0, 0, 0, R * 32, R * 32, S, S, S, (long)R * 32,
         1, 1, 0, 0, 0, 0, 0, 0, 1.f / S, 0, 0);
    k_opm_perm<<<g1(RR * 1024), 256, 0, stream>>>(LOG, ATT, R, 32);
    CVT_T(B.opm.o.w, 1024, CZ);
    gemm(stream, ATT, WB, z_cur, B.opm.o.b, z_cur, 0, (int)RR, CZ, 1024,
         1024, 1024, CZ, 1, 1, 0, 0, 0, 0, 0, 0, 1.f, 0, 0);

    // ================= Triangle multiplicative updates =================
    auto triMul = [&](const PTriMul& P, int swap) {
      LN(z_cur, ZLN, P.ln, RR, CZ);
      PROJ(ZLN, P.ag, CZ, CZ, RR, BUF1, 2, 0, 0, 0);
      PROJ(ZLN, P.ap, CZ, CZ, RR, BUF2, 0, 0, 0, 0);
      k_pack_gate<<<g1(RR * CZ), 256, 0, stream>>>(BUF1, BUF2, AT, R, CZ, swap);
      PROJ(ZLN, P.bg, CZ, CZ, RR, BUF1, 2, 0, 0, 0);
      PROJ(ZLN, P.bp, CZ, CZ, RR, BUF2, 0, 0, 0, 0);
      k_pack_gate<<<g1(RR * CZ), 256, 0, stream>>>(BUF1, BUF2, BT, R, CZ, swap);
      // x[c][i][j] = sum_k At[c][i][k]*Bt[c][j][k] ; 128 batched NT GEMMs
      gemm(stream, AT, BT, X1, 0, 0, 0, R, R, R, R, R, R,
           CZ, 1, RR, 0, RR, 0, RR, 0, 1.f, 0, 0);
      k_perm_cij_ijc<<<g1(RR * CZ), 256, 0, stream>>>(X1, X2, R, CZ);
      LN(X2, MLN, P.ln_out, RR, CZ);           // reuse MLN as bf16 scratch
      PROJ(ZLN, P.g, CZ, CZ, RR, GMAT, 2, 0, 0, 0);
      CVT_T(P.o.w, CZ, CZ);
      gemm(stream, MLN, WB, z_cur, P.o.b, z_cur, GMAT, (int)RR, CZ, CZ,
           CZ, CZ, CZ, 1, 1, 0, 0, 0, 0, 0, 0, 1.f, 0, 0);
    };
    triMul(B.tmo, 0);   // outgoing: ikc,jkc
    triMul(B.tmi, 1);   // incoming: kic,kjc

    // ================= Triangle attention =================
    auto triAtt = [&](const PAttn& P, int starting) {
      LN(z_cur, ZLN, P.ln, RR, CZ);
      PROJ(ZLN, P.q, CZ, CZ, RR, QB, 0, 1, 0, 0);
      PROJ(ZLN, P.k, CZ, CZ, RR, KB, 0, 1, 0, 0);
      PROJ(ZLN, P.v, CZ, CZ, RR, VB, 0, 1, 0, 0);
      PROJ(ZLN, P.g, CZ, CZ, RR, GT, 2, 0, 0, 0);
      PROJ(ZLN, P.bias, CZ, HT, RR, PB, 0, 0, 0, 0);
      k_vperm<<<g1(RR * CZ), 256, 0, stream>>>(VB, VT, R, R, HT, CH, starting ? 0 : 1);
      if (starting) {
        // logits[h,i,j,k]: batch (i,h), rows j, cols k ; bias[j,k,h]
        gemm(stream, QB, KB, LOG, 0, 0, 0, R, R, CH, CZ, CZ, R,
             R * HT, HT, (long)R * CZ, CH, (long)R * CZ, CH,
             (long)HT * RR, RR, rsq32, 0, 0);
        k_softmax<<<dim3((unsigned)(R * HT * R)), 128, 0, stream>>>(
            LOG, ATT, PB, R, R, HT, 1L, (long)R * HT, (long)HT);
      } else {
        // logits[h,i,j,k]: batch (j,h), rows i, cols k ; bias[k,i,h]
        gemm(stream, QB, KB, LOG, 0, 0, 0, R, R, CH, (long)R * CZ, (long)R * CZ, R,
             R * HT, HT, CZ, CH, CZ, CH, (long)HT * RR, RR, rsq32, 0, 0);
        k_softmax<<<dim3((unsigned)(R * HT * R)), 128, 0, stream>>>(
            LOG, ATT, PB, R, R, HT, 1L, (long)HT, (long)R * HT);
      }
      gemm(stream, ATT, VT, BUF1, 0, 0, 0, R, CH, R, R, R, CH,
           R * HT, HT, (long)HT * RR, RR, (long)HT * CH * R, (long)CH * R,
           (long)HT * R * CH, (long)R * CH, 1.f, 0, 0);
      k_gate_mul<<<g1(RR * CZ), 256, 0, stream>>>(BUF1, GT, GOB, R, R, HT, CH,
                                                  starting ? 0 : 1);
      CVT_T(P.o.w, CZ, CZ);
      gemm(stream, GOB, WB, z_cur, P.o.b, z_cur, 0, (int)RR, CZ, CZ,
           CZ, CZ, CZ, 1, 1, 0, 0, 0, 0, 0, 0, 1.f, 0, 0);
    };
    triAtt(B.tas, 1);
    triAtt(B.tae, 0);

    // ================= Pair transition =================
    LN(z_cur, ZLN, B.pt.ln, RR, CZ);
    CVT_T(B.pt.w1.w, CZ, 4 * CZ);
    gemm(stream, ZLN, WB, H1, B.pt.w1.b, 0, 0, (int)RR, 4 * CZ, CZ,
         CZ, CZ, 4 * CZ, 1, 1, 0, 0, 0, 0, 0, 0, 1.f, 1, 1);
    CVT_T(B.pt.w2.w, 4 * CZ, CZ);
    gemm(stream, H1, WB, z_cur, B.pt.w2.b, z_cur, 0, (int)RR, CZ, 4 * CZ,
         4 * CZ, 4 * CZ, CZ, 1, 1, 0, 0, 0, 0, 0, 0, 1.f, 0, 0);
  }

  // ================= Final single-representation head =================
  k_cvt<<<g1((long)R * CM), 256, 0, stream>>>(m_cur, MLN, (long)R * CM);
  CVT_T(fin.w, CM, CS);
  gemm(stream, MLN, WB, s_out, fin.b, 0, 0, R, CS, CM, CM, CM, CS,
       1, 1, 0, 0, 0, 0, 0, 0, 1.f, 0, 0);
}
